// HyperbolicUniform_27006754357262
// MI455X (gfx1250) — matrix operand order repaired and checked by reading, not compiled, not run on hardware
//
#include <hip/hip_runtime.h>
#include <hip/hip_bf16.h>
#include <stdint.h>

// ---------------------------------------------------------------------------
// HyperbolicUniform for MI455X (gfx1250, wave32, WMMA).
//
//   h1 = gelu(x @ W1 + b1)        [4096,256]   WMMA f16 -> f32 acc
//   h2 = gelu(h1 @ W2 + b2)       [4096,256]
//   u  = h2 @ W3 + b3             [4096,128]   fp32
//   hyp = expmap0(u)              [4096,128]   fp32 (output #2) + f16 copy
//   logits = -poincare_dist(hyp @ protos^T)/0.07   [4096,16384]
//
// Dist kernel: block = 4 waves, block tile 128(M) x 64(N).
//   - 64x128 f16 prototype tile staged in LDS via GLOBAL_LOAD_ASYNC_TO_LDS_B128
//   - each wave: 32x64 register tile = 8 f32 accumulators,
//     2 A-frags x 4 B-frags per k-step -> 0.75 operand loads per WMMA
//   - epilogue uses hardware TRANS ops (v_sqrt/v_rcp/v_log) instead of
//     precise libm sequences: ~1 ulp, dwarfed by the f16 GEMM rounding.
// ---------------------------------------------------------------------------

typedef __attribute__((ext_vector_type(16))) _Float16 v16h;
typedef __attribute__((ext_vector_type(8)))  _Float16 v8h;
typedef __attribute__((ext_vector_type(8)))  float    v8f;

#define B_SZ   4096
#define F_IN   768
#define H_SZ   256
#define D_SZ   128
#define N_SZ   16384

constexpr float kCC      = 3.0f;
constexpr float kSQRTC   = 1.7320508075688772f;
constexpr float kMINNORM = 1e-15f;
constexpr float kEPS     = 1e-5f;
constexpr float kINVTEMP = 14.285714285714286f;   // 1/0.07
constexpr float kLN2     = 0.69314718055994531f;
// logits = -(1/sqrt(c)) * ln(ratio) / temp = kPOST * log2(ratio)
constexpr float kPOST    = -(kINVTEMP * kLN2 / kSQRTC);

// ---------------- WMMA fragment loaders (contiguous k order, same in A and B) -----------

// A-matrix 16x32 f16 (MxK), row-major source, leading dim lda (halves):
//   lanes 0-15 : M=lane,    K = k0+0..15
//   lanes 16-31: M=lane-16, K = k0+16..31
__device__ inline v16h load_A_frag(const _Float16* __restrict__ A, int lda,
                                   int m0, int k0, int lane) {
  const int mrow = lane & 15;
  const int hsel = (lane >> 4) * 16;
  const _Float16* p = A + (size_t)(m0 + mrow) * lda + k0 + hsel;
  return *(const v16h*)(p);
}

// B-matrix 32x16 f16 (KxN) from transposed operand Bt[N][K] (global):
//   lanes 0-15 : N=lane,    K = k0+0..15  (16 contiguous halves)
//   lanes 16-31: N=lane-16, K = k0+16..31
__device__ inline v16h load_B_frag(const _Float16* __restrict__ Bt, int ldb,
                                   int n0, int k0, int lane) {
  const int ncol = lane & 15;
  const int koff = (lane >> 4) * 16;
  return *(const v16h*)(Bt + (size_t)(n0 + ncol) * ldb + k0 + koff);
}

// Same B layout but from an LDS tile sB[64][128] (halves).
__device__ inline v16h load_B_frag_lds(const _Float16* sB, int nt16,
                                       int k0, int lane) {
  const int ncol = lane & 15;
  const int koff = (lane >> 4) * 16;
  return *(const v16h*)(sB + (nt16 + ncol) * D_SZ + k0 + koff);
}

__device__ inline float gelu_exact(float v) {
  return 0.5f * v * (1.0f + erff(v * 0.70710678118654752f));
}

// 16-byte async global->LDS copy (gfx1250, ASYNCcnt-tracked).
__device__ inline void async_copy_b128(uint32_t lds_addr, const void* gptr) {
  asm volatile("global_load_async_to_lds_b128 %0, %1, off"
               :: "v"(lds_addr), "v"((uint64_t)(uintptr_t)gptr)
               : "memory");
}

// ---------------- prep kernels ---------------------------------------------

__global__ void cvt_f32_f16(const float* __restrict__ in,
                            _Float16* __restrict__ out, int n) {
  int i = blockIdx.x * blockDim.x + threadIdx.x;
  if (i < n) out[i] = (_Float16)in[i];
}

// in: [rows][cols] f32 row-major  ->  out: [cols][rows] f16 row-major
__global__ void transpose_cvt(const float* __restrict__ in,
                              _Float16* __restrict__ out, int rows, int cols) {
  int c = blockIdx.x * 16 + threadIdx.x;
  int r = blockIdx.y * 16 + threadIdx.y;
  if (c < cols && r < rows)
    out[(size_t)c * rows + r] = (_Float16)in[(size_t)r * cols + c];
}

__global__ void rownorm2_kernel(const float* __restrict__ p,
                                float* __restrict__ An, int rows) {
  int r = blockIdx.x * blockDim.x + threadIdx.x;
  if (r >= rows) return;
  const float4* row = (const float4*)(p + (size_t)r * D_SZ);
  float s = 0.f;
#pragma unroll 8
  for (int d = 0; d < D_SZ / 4; ++d) {
    float4 v = row[d];
    s += v.x * v.x + v.y * v.y + v.z * v.z + v.w * v.w;
  }
  An[r] = s;
}

// ---------------- MLP GEMM: one wave per 16x16 tile ------------------------

template <bool GELU_F16OUT>
__global__ void wmma_mlp_gemm(const _Float16* __restrict__ A,
                              const _Float16* __restrict__ Bt,
                              const float* __restrict__ bias,
                              void* __restrict__ outv,
                              int M, int N, int K) {
  const int lane = threadIdx.x & 31;
  const int wave = threadIdx.x >> 5;
  const int n0 = blockIdx.x * 16;
  const int m0 = (blockIdx.y * 4 + wave) * 16;

  v8f acc = {};
  for (int k0 = 0; k0 < K; k0 += 32) {
    v16h a = load_A_frag(A, K, m0, k0, lane);
    v16h b = load_B_frag(Bt, K, n0, k0, lane);
    acc = __builtin_amdgcn_wmma_f32_16x16x32_f16(false, a, false, b,
                                                 (short)0, acc, false, false);
  }

  const int n  = lane & 15;
  const int mb = (lane >> 4) * 8;
  const float bv = bias[n0 + n];
#pragma unroll
  for (int r = 0; r < 8; ++r) {
    float v = acc[r] + bv;
    if (GELU_F16OUT) {
      v = gelu_exact(v);
      ((_Float16*)outv)[(size_t)(m0 + mb + r) * N + n0 + n] = (_Float16)v;
    } else {
      ((float*)outv)[(size_t)(m0 + mb + r) * N + n0 + n] = v;
    }
  }
}

// ---------------- expmap0: one block (128 threads) per row -----------------

__global__ void expmap0_kernel(const float* __restrict__ u,
                               float* __restrict__ hyp_f32,
                               _Float16* __restrict__ hyp_f16,
                               float* __restrict__ Bn) {
  const int row = blockIdx.x;
  const int t   = threadIdx.x;          // 0..127
  float v = u[(size_t)row * D_SZ + t];
  float s = v * v;
#pragma unroll
  for (int o = 16; o > 0; o >>= 1) s += __shfl_xor(s, o, 32);
  __shared__ float red[4];
  if ((t & 31) == 0) red[t >> 5] = s;
  __syncthreads();
  float total = red[0] + red[1] + red[2] + red[3];
  float norm  = fmaxf(sqrtf(total), kMINNORM);
  float th    = tanhf(kSQRTC * norm);
  float scale = th / (kSQRTC * norm);
  float h = scale * v;
  hyp_f32[(size_t)row * D_SZ + t] = h;
  hyp_f16[(size_t)row * D_SZ + t] = (_Float16)h;
  if (t == 0) Bn[row] = (th * th) / (kSQRTC * kSQRTC); // ||hyp||^2
}

// ---------------- fused distance GEMM + Poincare epilogue ------------------
// Block: 128 threads (4 waves). Block tile: 128 (M) x 64 (N).
// LDS: 64x128 f16 prototype tile, async-staged, shared by all 4 waves.
// Wave tile: 32 (M) x 64 (N) -> 8 f32 accumulators.

__global__ void __launch_bounds__(128)
dist_logits_kernel(const _Float16* __restrict__ hyp,   // [B][128]
                   const _Float16* __restrict__ proto, // [N][128]
                   const float* __restrict__ Bn,       // [B]
                   const float* __restrict__ An,       // [N]
                   float* __restrict__ logits) {       // [B][N]
  __shared__ __align__(32) _Float16 sB[64 * D_SZ];     // 16 KB

  const int lane = threadIdx.x & 31;
  const int wave = threadIdx.x >> 5;
  const int n0 = blockIdx.x * 64;               // 4 n-tiles of 16
  const int m0 = blockIdx.y * 128 + wave * 32;  // 2 m-tiles of 16 per wave

  // ---- async-stage proto[n0 .. n0+63][0..127] into LDS (16 KB) ----
  {
    const char* gsrc = (const char*)(proto + (size_t)n0 * D_SZ);
    uint32_t lds_base = (uint32_t)(uintptr_t)(&sB[0]);
#pragma unroll
    for (int i = 0; i < 8; ++i) {
      int chunk = i * 128 + threadIdx.x;        // 1024 x 16B chunks
      async_copy_b128(lds_base + chunk * 16, gsrc + (size_t)chunk * 16);
    }
    asm volatile("s_wait_asynccnt 0x0" ::: "memory");
    __syncthreads();
  }

  // ---- 32x64 register tile: acc[mi][ni] ----
  v8f acc[2][4];
#pragma unroll
  for (int mi = 0; mi < 2; ++mi)
#pragma unroll
    for (int ni = 0; ni < 4; ++ni) acc[mi][ni] = {};

#pragma unroll
  for (int k0 = 0; k0 < D_SZ; k0 += 32) {
    v16h a0 = load_A_frag(hyp, D_SZ, m0,      k0, lane);
    v16h a1 = load_A_frag(hyp, D_SZ, m0 + 16, k0, lane);
    v16h b0 = load_B_frag_lds(sB,  0, k0, lane);
    v16h b1 = load_B_frag_lds(sB, 16, k0, lane);
    v16h b2 = load_B_frag_lds(sB, 32, k0, lane);
    v16h b3 = load_B_frag_lds(sB, 48, k0, lane);
    acc[0][0] = __builtin_amdgcn_wmma_f32_16x16x32_f16(false, a0, false, b0, (short)0, acc[0][0], false, false);
    acc[0][1] = __builtin_amdgcn_wmma_f32_16x16x32_f16(false, a0, false, b1, (short)0, acc[0][1], false, false);
    acc[0][2] = __builtin_amdgcn_wmma_f32_16x16x32_f16(false, a0, false, b2, (short)0, acc[0][2], false, false);
    acc[0][3] = __builtin_amdgcn_wmma_f32_16x16x32_f16(false, a0, false, b3, (short)0, acc[0][3], false, false);
    acc[1][0] = __builtin_amdgcn_wmma_f32_16x16x32_f16(false, a1, false, b0, (short)0, acc[1][0], false, false);
    acc[1][1] = __builtin_amdgcn_wmma_f32_16x16x32_f16(false, a1, false, b1, (short)0, acc[1][1], false, false);
    acc[1][2] = __builtin_amdgcn_wmma_f32_16x16x32_f16(false, a1, false, b2, (short)0, acc[1][2], false, false);
    acc[1][3] = __builtin_amdgcn_wmma_f32_16x16x32_f16(false, a1, false, b3, (short)0, acc[1][3], false, false);
  }

  // ---- Poincare-distance epilogue (hardware TRANS ops) ----
  const int n  = lane & 15;
  const int mb = (lane >> 4) * 8;
#pragma unroll
  for (int ni = 0; ni < 4; ++ni) {
    const int nc = n0 + ni * 16 + n;
    const float Ann  = An[nc];
    const float beta = 1.0f - kCC * Ann;
#pragma unroll
    for (int mi = 0; mi < 2; ++mi) {
#pragma unroll
      for (int r = 0; r < 8; ++r) {
        const int mr = m0 + mi * 16 + mb + r;
        float P  = acc[mi][ni][r];
        float Bm = Bn[mr];
        float alpha  = 1.0f - 2.0f * kCC * P + kCC * Bm;
        float denom  = fmaxf(1.0f - 2.0f * kCC * P + kCC * kCC * Ann * Bm, kMINNORM);
        float num_sq = alpha * alpha * Ann + beta * beta * Bm - 2.0f * alpha * beta * P;
        float mnorm  = __builtin_amdgcn_sqrtf(fmaxf(num_sq, kMINNORM)) *
                       __builtin_amdgcn_rcpf(denom);
        float arg    = fminf(kSQRTC * mnorm, 1.0f - kEPS);
        // artanh(x) = 0.5*ln((1+x)/(1-x)); fold ln2, 2/sqrt(c), 1/temp -> kPOST
        float ratio  = (1.0f + arg) * __builtin_amdgcn_rcpf(1.0f - arg);
        logits[(size_t)mr * N_SZ + nc] = kPOST * __builtin_amdgcn_logf(ratio);
      }
    }
  }
}

// ---------------------------------------------------------------------------

extern "C" void kernel_launch(void* const* d_in, const int* in_sizes, int n_in,
                              void* d_out, int out_size, void* d_ws, size_t ws_size,
                              hipStream_t stream) {
  const float* x   = (const float*)d_in[0];
  const float* W1  = (const float*)d_in[1];
  const float* b1  = (const float*)d_in[2];
  const float* W2  = (const float*)d_in[3];
  const float* b2  = (const float*)d_in[4];
  const float* W3  = (const float*)d_in[5];
  const float* b3  = (const float*)d_in[6];
  const float* prt = (const float*)d_in[7];
  float* out = (float*)d_out;

  // ---- carve workspace (deterministic offsets, 256B aligned) ----
  char* base = (char*)d_ws;
  size_t off = 0;
  auto carve = [&](size_t bytes) -> char* {
    char* p = base + off;
    off += (bytes + 255) & ~(size_t)255;
    return p;
  };
  _Float16* x_h    = (_Float16*)carve((size_t)B_SZ * F_IN * 2);
  _Float16* W1t    = (_Float16*)carve((size_t)F_IN * H_SZ * 2);
  _Float16* W2t    = (_Float16*)carve((size_t)H_SZ * H_SZ * 2);
  _Float16* W3t    = (_Float16*)carve((size_t)H_SZ * D_SZ * 2);
  _Float16* prt_h  = (_Float16*)carve((size_t)N_SZ * D_SZ * 2);
  _Float16* h1     = (_Float16*)carve((size_t)B_SZ * H_SZ * 2);
  _Float16* h2     = (_Float16*)carve((size_t)B_SZ * H_SZ * 2);
  float*    u      = (float*)   carve((size_t)B_SZ * D_SZ * 4);
  _Float16* hyp_h  = (_Float16*)carve((size_t)B_SZ * D_SZ * 2);
  float*    Bn     = (float*)   carve((size_t)B_SZ * 4);
  float*    An     = (float*)   carve((size_t)N_SZ * 4);

  // ---- prep ----
  {
    int n = B_SZ * F_IN;
    cvt_f32_f16<<<(n + 255) / 256, 256, 0, stream>>>(x, x_h, n);
  }
  {
    dim3 b(16, 16);
    transpose_cvt<<<dim3(H_SZ / 16, F_IN / 16), b, 0, stream>>>(W1, W1t, F_IN, H_SZ);
    transpose_cvt<<<dim3(H_SZ / 16, H_SZ / 16), b, 0, stream>>>(W2, W2t, H_SZ, H_SZ);
    transpose_cvt<<<dim3(D_SZ / 16, H_SZ / 16), b, 0, stream>>>(W3, W3t, H_SZ, D_SZ);
  }
  {
    int n = N_SZ * D_SZ;
    cvt_f32_f16<<<(n + 255) / 256, 256, 0, stream>>>(prt, prt_h, n);
    rownorm2_kernel<<<(N_SZ + 255) / 256, 256, 0, stream>>>(prt, An, N_SZ);
  }

  // ---- MLP ----
  wmma_mlp_gemm<true><<<dim3(H_SZ / 16, B_SZ / 64), 128, 0, stream>>>(
      x_h, W1t, b1, (void*)h1, B_SZ, H_SZ, F_IN);
  wmma_mlp_gemm<true><<<dim3(H_SZ / 16, B_SZ / 64), 128, 0, stream>>>(
      h1, W2t, b2, (void*)h2, B_SZ, H_SZ, H_SZ);
  wmma_mlp_gemm<false><<<dim3(D_SZ / 16, B_SZ / 64), 128, 0, stream>>>(
      h2, W3t, b3, (void*)u, B_SZ, D_SZ, H_SZ);

  // ---- expmap0 ----
  float* hyp_out = out + (size_t)B_SZ * N_SZ;
  expmap0_kernel<<<B_SZ, 128, 0, stream>>>(u, hyp_out, hyp_h, Bn);

  // ---- fused distance GEMM + hyperbolic epilogue ----
  dist_logits_kernel<<<dim3(N_SZ / 64, B_SZ / 128), 128, 0, stream>>>(
      hyp_h, prt_h, Bn, An, out);
}
